// MultiHeadAttention_41042707480980
// MI455X (gfx1250) — matrix operand order repaired and checked by reading, not compiled, not run on hardware
//
#include <hip/hip_runtime.h>
#include <hip/hip_bf16.h>

// ---------------------------------------------------------------------------
// MI455X (gfx1250) multi-head attention forward, bf16 WMMA everywhere.
//   x:[2,2048,2048] f32, Wq/Wk/Wv/Wo:[2048,2048] f32 ([out,in])
//   d_out = out(8388608 f32) | k(8388608 f32, [B,H,S,dh]) | v(8388608 f32)
// ---------------------------------------------------------------------------

typedef __bf16 bf16;
typedef __attribute__((ext_vector_type(16))) __bf16 v16bf;
typedef __attribute__((ext_vector_type(8)))  float  v8f;

#define DMODEL 2048
#define NHEADS 16
#define DHEAD  128
#define BATCH  2
#define SEQ    2048
#define MROWS  (BATCH * SEQ)   // 4096

// CDNA5 async global->LDS path (ASYNCcnt-tracked, no VGPR round trip).
// Prototype discovered from compiler diagnostics:
//   void(v4i addrspace(1)*, v4i addrspace(3)*, imm int offset, imm int cpol)
#if defined(__has_builtin)
#if __has_builtin(__builtin_amdgcn_global_load_async_to_lds_b128) && \
    __has_builtin(__builtin_amdgcn_s_wait_asynccnt)
#define USE_ASYNC_LDS 1
#endif
#endif

#if defined(USE_ASYNC_LDS)
typedef int v4i_gcc __attribute__((vector_size(16)));
typedef __attribute__((address_space(1))) v4i_gcc* gptr_v4i;
typedef __attribute__((address_space(3))) v4i_gcc* lptr_v4i;
#endif

__device__ __forceinline__ v8f wmma_bf16(v16bf a, v16bf b, v8f c) {
    // (neg_a, A, neg_b, B, c_mod, C, reuse_a, reuse_b)
    return __builtin_amdgcn_wmma_f32_16x16x32_bf16(false, a, false, b, (short)0, c, false, false);
}

// A fragment: 16x32 bf16, row-major source with leading dim `ld` (elements).
// Layout (ISA 7.12.2): lanes 0-15 row=lane, K in {0..7,16..23}; lanes 16-31
// row=lane-16, K in {8..15,24..31}; VGPR j packs 2 consecutive K.
__device__ __forceinline__ v16bf load_a_frag(const bf16* __restrict__ p, int ld, int lane) {
    int row = lane & 15;
    int kh  = (lane >> 4) << 3;          // 0 or 8
    const bf16* src = p + (size_t)row * ld;
    v16bf a;
#pragma unroll
    for (int j = 0; j < 8; ++j) a[j]     = src[kh + j];        // K = kh..kh+7
#pragma unroll
    for (int j = 0; j < 8; ++j) a[8 + j] = src[16 + kh + j];   // K = 16+kh..
    return a;
}

// B fragment: 32x16 bf16 where B[k][n] = src[n*ld + k] (i.e. operand is the
// transpose of a row-major [16 x 32] slab).
// Lanes 0-15: col=lane, K in {0..7,16..23}; lanes 16-31: col=lane-16, K in {8..15,24..31}.
__device__ __forceinline__ v16bf load_b_frag(const bf16* __restrict__ p, int ld, int lane) {
    int n    = lane & 15;
    int kh   = (lane >> 4) << 3;         // 0 or 8
    const bf16* src = p + (size_t)n * ld;
    v16bf b;
#pragma unroll
    for (int j = 0; j < 8; ++j) b[j]     = src[kh + j];
#pragma unroll
    for (int j = 0; j < 8; ++j) b[8 + j] = src[16 + kh + j];
    return b;
}

// ---------------------------------------------------------------------------
__global__ void f32_to_bf16_kernel(const float* __restrict__ in,
                                   bf16* __restrict__ out, int n) {
    int i = blockIdx.x * blockDim.x + threadIdx.x;
    if (i < n) out[i] = (bf16)in[i];
}

// ---------------------------------------------------------------------------
// GEMM: C[M,N] = A[M,K] * W[N,K]^T   (M=4096, N=2048, K=2048)
// Block tile 128x128, K-step 32. 8 waves in a 4x2 grid, each wave owns a
// 32x64 slab: 2 A-frags x 4 B-frags = 8 WMMA per K-step (4x operand reuse).
// mode 0: f32 row-major [M,2048] -> Cf32                   (out projection)
// mode 1: bf16 [B,H,S,dh]        -> C16                    (Q)
// mode 2: bf16 [B,H,S,dh] -> C16 ; f32 [B,H,S,dh] -> Cf32  (K)
// mode 3: bf16 [B,H,dh,S] -> C16 ; f32 [B,H,S,dh] -> Cf32  (V, bf16 transposed)
// ---------------------------------------------------------------------------
__device__ __forceinline__ void store_c_elem(float v, int m, int n, int mode,
                                             float* __restrict__ Cf32,
                                             bf16* __restrict__ C16) {
    if (mode == 0) { Cf32[(size_t)m * DMODEL + n] = v; return; }
    int b = m >> 11, s = m & (SEQ - 1);
    int h = n >> 7, d = n & (DHEAD - 1);
    size_t head = ((size_t)(b * NHEADS + h) * SEQ + s) * DHEAD + d;
    if (mode == 1) { C16[head] = (bf16)v; }
    else if (mode == 2) { C16[head] = (bf16)v; Cf32[head] = v; }
    else { // mode 3
        size_t t = ((size_t)(b * NHEADS + h) * DHEAD + d) * SEQ + s;
        C16[t] = (bf16)v; Cf32[head] = v;
    }
}

#define GTM 128
#define GTN 128
#define GTK 32

__global__ __launch_bounds__(256) void gemm_wmma_kernel(
        const bf16* __restrict__ A,   // [M,K] row-major
        const bf16* __restrict__ W,   // [N,K] row-major
        float* __restrict__ Cf32, bf16* __restrict__ C16,
        int M, int N, int K, int mode) {
    __shared__ __align__(16) bf16 lA[GTM][GTK];   // 8 KB
    __shared__ __align__(16) bf16 lB[GTN][GTK];   // 8 KB

    int tid  = threadIdx.x;
    int lane = tid & 31;
    int wave = tid >> 5;
    int wm = wave >> 1;                // 0..3  -> rows wm*32
    int wn = wave & 1;                 // 0..1  -> cols wn*64
    int bm = blockIdx.x, bn = blockIdx.y;

    // cooperative copy: each thread owns a contiguous 32B (16 bf16) run
    int lrow = tid >> 1;               // 0..127
    int lk   = (tid & 1) << 4;         // 0 or 16
    const bf16* Arow = A + (size_t)(bm * GTM + lrow) * K + lk;
    const bf16* Wrow = W + (size_t)(bn * GTN + lrow) * K + lk;

    v8f c[2][4];
#pragma unroll
    for (int mi = 0; mi < 2; ++mi)
#pragma unroll
        for (int ni = 0; ni < 4; ++ni) c[mi][ni] = (v8f){};

    for (int k0 = 0; k0 < K; k0 += GTK) {
#if defined(USE_ASYNC_LDS)
        // GLOBAL_LOAD_ASYNC_TO_LDS_B128 x4: two 16B chunks for A, two for B.
        __builtin_amdgcn_global_load_async_to_lds_b128(
            (gptr_v4i)(Arow + k0),     (lptr_v4i)&lA[lrow][lk],     0, 0);
        __builtin_amdgcn_global_load_async_to_lds_b128(
            (gptr_v4i)(Arow + k0 + 8), (lptr_v4i)&lA[lrow][lk + 8], 0, 0);
        __builtin_amdgcn_global_load_async_to_lds_b128(
            (gptr_v4i)(Wrow + k0),     (lptr_v4i)&lB[lrow][lk],     0, 0);
        __builtin_amdgcn_global_load_async_to_lds_b128(
            (gptr_v4i)(Wrow + k0 + 8), (lptr_v4i)&lB[lrow][lk + 8], 0, 0);
        __builtin_amdgcn_s_wait_asynccnt(0);   // drain own ASYNCcnt pre-barrier
#else
        *(uint4*)&lA[lrow][lk]     = *(const uint4*)(Arow + k0);
        *(uint4*)&lA[lrow][lk + 8] = *(const uint4*)(Arow + k0 + 8);
        *(uint4*)&lB[lrow][lk]     = *(const uint4*)(Wrow + k0);
        *(uint4*)&lB[lrow][lk + 8] = *(const uint4*)(Wrow + k0 + 8);
#endif
        if (k0 + GTK < K) {            // pull next K slab toward near caches
            __builtin_prefetch(Arow + k0 + GTK, 0, 3);
            __builtin_prefetch(Wrow + k0 + GTK, 0, 3);
        }
        __syncthreads();

        v16bf af[2];
#pragma unroll
        for (int mi = 0; mi < 2; ++mi)
            af[mi] = load_a_frag(&lA[wm * 32 + mi * 16][0], GTK, lane);
#pragma unroll
        for (int ni = 0; ni < 4; ++ni) {
            v16bf bfr = load_b_frag(&lB[wn * 64 + ni * 16][0], GTK, lane);
#pragma unroll
            for (int mi = 0; mi < 2; ++mi)
                c[mi][ni] = wmma_bf16(af[mi], bfr, c[mi][ni]);
        }
        __syncthreads();
    }

    int r0 = bm * GTM + wm * 32 + ((lane >> 4) << 3);
    int n0 = bn * GTN + wn * 64 + (lane & 15);
#pragma unroll
    for (int mi = 0; mi < 2; ++mi)
#pragma unroll
        for (int ni = 0; ni < 4; ++ni)
#pragma unroll
            for (int r = 0; r < 8; ++r)
                store_c_elem(c[mi][ni][r], r0 + mi * 16 + r, n0 + ni * 16,
                             mode, Cf32, C16);
}

// ---------------------------------------------------------------------------
// Causal flash attention. One wave = one (b,h,16-query tile).
// Q16,K16: [B*H, S, dh] bf16.  V16t: [B*H, dh, S] bf16 (transposed).
// AO16 out: [B*S, D] bf16  (m = b*S+s, col = h*dh + d), feeds out-projection.
// ---------------------------------------------------------------------------
__global__ __launch_bounds__(256) void attn_wmma_kernel(
        const bf16* __restrict__ Q16, const bf16* __restrict__ K16,
        const bf16* __restrict__ V16t, bf16* __restrict__ AO16) {
    __shared__ __align__(16) bf16 lP[8][16 * 32];   // per-wave P tile

    int lane = threadIdx.x & 31;
    int wave = threadIdx.x >> 5;
    int bh = blockIdx.x >> 4;           // 0..31 = b*16+h
    int qg = blockIdx.x & 15;           // query-tile group
    int qt = qg * 8 + wave;             // 0..127
    int b  = bh >> 4, h = bh & 15;
    int qbase = qt * 16;

    const bf16* Qb = Q16 + (size_t)bh * SEQ * DHEAD;
    const bf16* Kb = K16 + (size_t)bh * SEQ * DHEAD;
    const bf16* Vb = V16t + (size_t)bh * DHEAD * SEQ;
    bf16* pP = &lP[wave][0];

    // Q tile A-fragments, dh chunks of 32, kept in registers.
    v16bf qa[4];
#pragma unroll
    for (int c = 0; c < 4; ++c)
        qa[c] = load_a_frag(Qb + (size_t)qbase * DHEAD + c * 32, DHEAD, lane);

    v8f acc[8];
#pragma unroll
    for (int f = 0; f < 8; ++f) acc[f] = (v8f){};
    float mrun[8], lrun[8];
#pragma unroll
    for (int r = 0; r < 8; ++r) { mrun[r] = -1e30f; lrun[r] = 0.0f; }

    const float sl2e = 0.08838834764831845f * 1.4426950408889634f; // 1/sqrt(dh)*log2(e)
    int rofs = (lane >> 4) << 3;
    int col  = lane & 15;
    int nsteps = ((qbase + 15) >> 5) + 1;

    for (int st = 0; st < nsteps; ++st) {
        int kbase = st * 32;
        // --- scores: Q(16x128) @ K^T -> two 16x16 frags ---
        v8f s0 = (v8f){}, s1 = (v8f){};
#pragma unroll
        for (int c = 0; c < 4; ++c) {
            v16bf kb0 = load_b_frag(Kb + (size_t)kbase * DHEAD + c * 32, DHEAD, lane);
            v16bf kb1 = load_b_frag(Kb + (size_t)(kbase + 16) * DHEAD + c * 32, DHEAD, lane);
            s0 = wmma_bf16(qa[c], kb0, s0);
            s1 = wmma_bf16(qa[c], kb1, s1);
        }
        // --- causal mask + online softmax (scaled log2 domain) ---
        float p0[8], p1[8], alpha[8];
#pragma unroll
        for (int r = 0; r < 8; ++r) {
            int qrow = qbase + rofs + r;
            float t0 = (kbase + col      <= qrow) ? s0[r] * sl2e : -1e30f;
            float t1 = (kbase + 16 + col <= qrow) ? s1[r] * sl2e : -1e30f;
            float mx = fmaxf(t0, t1);
            mx = fmaxf(mx, __shfl_xor(mx, 1, 32));
            mx = fmaxf(mx, __shfl_xor(mx, 2, 32));
            mx = fmaxf(mx, __shfl_xor(mx, 4, 32));
            mx = fmaxf(mx, __shfl_xor(mx, 8, 32));
            float mnew = fmaxf(mrun[r], mx);
            alpha[r] = exp2f(mrun[r] - mnew);
            p0[r] = exp2f(t0 - mnew);
            p1[r] = exp2f(t1 - mnew);
            float rs = p0[r] + p1[r];
            rs += __shfl_xor(rs, 1, 32);
            rs += __shfl_xor(rs, 2, 32);
            rs += __shfl_xor(rs, 4, 32);
            rs += __shfl_xor(rs, 8, 32);
            lrun[r] = lrun[r] * alpha[r] + rs;
            mrun[r] = mnew;
        }
#pragma unroll
        for (int f = 0; f < 8; ++f)
#pragma unroll
            for (int r = 0; r < 8; ++r) acc[f][r] *= alpha[r];
        // --- P (C layout) -> LDS -> A fragment (wave-synchronous, DScnt order)
#pragma unroll
        for (int r = 0; r < 8; ++r) {
            pP[(rofs + r) * 32 + col]      = (bf16)p0[r];
            pP[(rofs + r) * 32 + col + 16] = (bf16)p1[r];
        }
        __builtin_amdgcn_wave_barrier();
        v16bf pa = load_a_frag(pP, 32, lane);
        // --- O += P(16x32) @ V(32x128): 8 dh frags from transposed V ---
#pragma unroll
        for (int f = 0; f < 8; ++f) {
            v16bf vb = load_b_frag(Vb + (size_t)(f * 16) * SEQ + kbase, SEQ, lane);
            acc[f] = wmma_bf16(pa, vb, acc[f]);
        }
    }

    // normalize and emit bf16 [m, h*dh+d]
#pragma unroll
    for (int r = 0; r < 8; ++r) {
        float inv = 1.0f / lrun[r];
        size_t m = (size_t)b * SEQ + qbase + rofs + r;
#pragma unroll
        for (int f = 0; f < 8; ++f)
            AO16[m * DMODEL + h * DHEAD + f * 16 + col] = (bf16)(acc[f][r] * inv);
    }
}

// ---------------------------------------------------------------------------
extern "C" void kernel_launch(void* const* d_in, const int* in_sizes, int n_in,
                              void* d_out, int out_size, void* d_ws, size_t ws_size,
                              hipStream_t stream) {
    const float* x  = (const float*)d_in[0];
    const float* Wq = (const float*)d_in[1];
    const float* Wk = (const float*)d_in[2];
    const float* Wv = (const float*)d_in[3];
    const float* Wo = (const float*)d_in[4];

    float* out  = (float*)d_out;                       // [4096,2048]
    float* kout = out + (size_t)MROWS * DMODEL;        // [B,H,S,dh]
    float* vout = kout + (size_t)MROWS * DMODEL;       // [B,H,S,dh]

    // scratch layout (bf16 elements); total 112 MB
    bf16* X16  = (bf16*)d_ws;
    bf16* Wq16 = X16  + (size_t)MROWS * DMODEL;        //  8388608
    bf16* Wk16 = Wq16 + (size_t)DMODEL * DMODEL;
    bf16* Wv16 = Wk16 + (size_t)DMODEL * DMODEL;
    bf16* Wo16 = Wv16 + (size_t)DMODEL * DMODEL;
    bf16* Q16  = Wo16 + (size_t)DMODEL * DMODEL;
    bf16* K16  = Q16  + (size_t)MROWS * DMODEL;
    bf16* V16t = K16  + (size_t)MROWS * DMODEL;
    bf16* AO16 = V16t + (size_t)MROWS * DMODEL;

    const int nX = MROWS * DMODEL;                     // 8388608
    const int nW = DMODEL * DMODEL;                    // 4194304
    f32_to_bf16_kernel<<<(nX + 255) / 256, 256, 0, stream>>>(x,  X16,  nX);
    f32_to_bf16_kernel<<<(nW + 255) / 256, 256, 0, stream>>>(Wq, Wq16, nW);
    f32_to_bf16_kernel<<<(nW + 255) / 256, 256, 0, stream>>>(Wk, Wk16, nW);
    f32_to_bf16_kernel<<<(nW + 255) / 256, 256, 0, stream>>>(Wv, Wv16, nW);
    f32_to_bf16_kernel<<<(nW + 255) / 256, 256, 0, stream>>>(Wo, Wo16, nW);

    dim3 ggrid(MROWS / GTM, DMODEL / GTN);             // 32 x 16
    gemm_wmma_kernel<<<ggrid, 256, 0, stream>>>(X16, Wq16, nullptr, Q16,
                                                MROWS, DMODEL, DMODEL, 1);
    gemm_wmma_kernel<<<ggrid, 256, 0, stream>>>(X16, Wk16, kout, K16,
                                                MROWS, DMODEL, DMODEL, 2);
    gemm_wmma_kernel<<<ggrid, 256, 0, stream>>>(X16, Wv16, vout, V16t,
                                                MROWS, DMODEL, DMODEL, 3);

    attn_wmma_kernel<<<BATCH * NHEADS * 16, 256, 0, stream>>>(Q16, K16, V16t, AO16);

    gemm_wmma_kernel<<<ggrid, 256, 0, stream>>>(AO16, Wo16, out, nullptr,
                                                MROWS, DMODEL, DMODEL, 0);
}
